// RGCN_48043504173161
// MI455X (gfx1250) — matrix operand order repaired and checked by reading, not compiled, not run on hardware
//
#include <hip/hip_runtime.h>
#include <hip/hip_bf16.h>

// CDNA5 (gfx1250) wave32 WMMA types
typedef __attribute__((ext_vector_type(16))) __bf16 v16bf;
typedef __attribute__((ext_vector_type(8)))  __bf16 v8bf;
typedef __attribute__((ext_vector_type(8)))  float  v8f;

#define RREL 8  // number of relations (root transform appended as relation 8)

// ---------------------------------------------------------------------------
// Edge degree count per (dst, relation): integer atomics
// ---------------------------------------------------------------------------
__global__ __launch_bounds__(256) void edge_count_kernel(
    const int* __restrict__ ei, const int* __restrict__ et,
    unsigned* __restrict__ cnt, int nE)
{
    int e = blockIdx.x * 256 + threadIdx.x;
    if (e >= nE) return;
    int dst = ei[nE + e];
    int r   = et[e];
    atomicAdd(&cnt[dst * RREL + r], 1u);
}

// In-place: uint count -> float 1/max(count,1)
__global__ __launch_bounds__(256) void invnorm_kernel(float* __restrict__ cnt, int n)
{
    int i = blockIdx.x * 256 + threadIdx.x;
    if (i >= n) return;
    unsigned c = reinterpret_cast<unsigned*>(cnt)[i];
    cnt[i] = 1.0f / (float)(c > 1u ? c : 1u);
}

// ---------------------------------------------------------------------------
// Per-relation GEMM: H[r] = X @ W_r  (r = 0..7 -> w, r = 8 -> root)
// X: [nrows, in_dim] f32 ; W: [R, in_dim, 128] f32 ; H: [9, nrows, 128] bf16
// Block: 256 thr (8 waves), 64 rows x 128 cols per block.
// Wave w: M-strip (w&3)*16, col half (w>>2)*64 -> 4 wmma tiles sharing A frag.
// Register-double-buffered staging: next K-step's global loads are issued
// right after the staging barrier and consumed by stage() one iteration later,
// hiding HBM latency behind ds_load + v_wmma.
// ---------------------------------------------------------------------------
__global__ __launch_bounds__(256) void rgemm_bf16_kernel(
    const float* __restrict__ X, const float* __restrict__ W,
    const float* __restrict__ root, __bf16* __restrict__ H,
    int nrows, int in_dim)
{
    __shared__ __bf16 As[64][40];   // [m][k]  (pad 40: 80B row stride, 16B aligned)
    __shared__ __bf16 Bt[128][40];  // [n][k]  weight slice, transposed

    const int r    = blockIdx.y;                       // 0..8
    const float* Wm = (r < RREL) ? (W + (size_t)r * in_dim * 128) : root;
    const int row0 = blockIdx.x * 64;
    const int tid  = threadIdx.x;
    const int lane = tid & 31;
    const int wave = tid >> 5;
    const int p    = lane & 15;      // row/col within 16-lane half
    const int half = lane >> 4;      // 0 or 1
    const int mt   = wave & 3;       // M sub-tile 0..3 (16 rows each)
    const int nb   = (wave >> 2) * 64;  // col base: 0 or 64

    // --- staging ownership ---
    // A tile: 64 rows x 32 k ; thread owns 8 contiguous k of one row
    const int sr   = tid >> 2;           // 0..63
    const int skg  = (tid & 3) * 8;      // 0,8,16,24
    const bool avalid = (row0 + sr) < nrows;
    const float* aptr = X + (size_t)(row0 + sr) * in_dim + skg;
    // W slice: 32 k x 128 c ; thread owns 2 items of (c fixed, 8 k's)
    const int sc   = tid & 127;          // column
    const int wkg0 = (tid >> 7) * 8;     // 0 or 8
    const int wkg1 = wkg0 + 16;          // 16 or 24

    float af[8], wf0[8], wf1[8];

    auto preload = [&](int kb) {
        #pragma unroll
        for (int i = 0; i < 8; ++i)
            af[i] = avalid ? aptr[kb + i] : 0.0f;       // 2x b128 loads
        #pragma unroll
        for (int j = 0; j < 8; ++j) {                   // strided b32 loads,
            wf0[j] = Wm[(size_t)(kb + wkg0 + j) * 128 + sc];  // all in flight
            wf1[j] = Wm[(size_t)(kb + wkg1 + j) * 128 + sc];
        }
    };

    auto stage = [&]() {
        v8bf av, w0v, w1v;
        #pragma unroll
        for (int i = 0; i < 8; ++i) {
            av[i]  = (__bf16)af[i];
            w0v[i] = (__bf16)wf0[i];
            w1v[i] = (__bf16)wf1[i];
        }
        *reinterpret_cast<v8bf*>(&As[sr][skg])  = av;   // ds_store_b128
        *reinterpret_cast<v8bf*>(&Bt[sc][wkg0]) = w0v;
        *reinterpret_cast<v8bf*>(&Bt[sc][wkg1]) = w1v;
    };

    v8f acc[4] = {v8f{}, v8f{}, v8f{}, v8f{}};

    preload(0);
    for (int k0 = 0; k0 < in_dim; k0 += 32) {
        stage();
        __syncthreads();
        if (k0 + 32 < in_dim) preload(k0 + 32);  // overlap with wmma below

        // A fragment (16x32 bf16, ISA 7.12.2): lane p = row M, element i:
        //   K = i + 8*half  (i<8) ;  K = i + 8 + 8*half  (i>=8)
        const int fr = mt * 16 + p;
        v16bf a;
        #pragma unroll
        for (int i = 0; i < 8; ++i)  a[i] = As[fr][8 * half + i];
        #pragma unroll
        for (int i = 8; i < 16; ++i) a[i] = As[fr][8 + 8 * half + i];

        // B fragments (32x16 bf16): lane p = col N, element i: K = i + 8*half (i<8) ; K = i + 8 + 8*half (i>=8)
        v16bf b[4];
        #pragma unroll
        for (int nt = 0; nt < 4; ++nt) {
            const __bf16* bc = &Bt[nb + nt * 16 + p][8 * half];
            #pragma unroll
            for (int i = 0; i < 16; ++i) b[nt][i] = bc[i < 8 ? i : i + 8];
        }

        #pragma unroll
        for (int nt = 0; nt < 4; ++nt) {
            acc[nt] = __builtin_amdgcn_wmma_f32_16x16x32_bf16(
                false, a, false, b[nt], (short)0, acc[nt], false, false);
        }
        __syncthreads();
    }

    // C layout: VGPR j -> M = j + 8*half, N = p
    #pragma unroll
    for (int nt = 0; nt < 4; ++nt) {
        #pragma unroll
        for (int j = 0; j < 8; ++j) {
            int grow = row0 + mt * 16 + j + 8 * half;
            if (grow < nrows) {
                int gcol = nb + nt * 16 + p;
                H[((size_t)r * nrows + grow) * 128 + gcol] = (__bf16)acc[nt][j];
            }
        }
    }
}

// ---------------------------------------------------------------------------
// Edge scatter: one wave per edge; 32 lanes x 4 f32 = 128-wide message.
// agg[dst] += h_all[rel][src] * (1/count(dst,rel))
// ---------------------------------------------------------------------------
__global__ __launch_bounds__(256) void edge_scatter_kernel(
    const int* __restrict__ ei, const int* __restrict__ et,
    const float* __restrict__ inv, const __bf16* __restrict__ H,
    float* __restrict__ agg, int nE, int nrows)
{
    int gid  = blockIdx.x * 256 + threadIdx.x;
    int e    = gid >> 5;
    if (e >= nE) return;
    int lane = gid & 31;
    int src  = ei[e];
    int dst  = ei[nE + e];
    int r    = et[e];
    float w  = inv[dst * RREL + r];

    const unsigned short* hp = reinterpret_cast<const unsigned short*>(
        H + ((size_t)r * nrows + src) * 128 + lane * 4);
    ushort4 raw = *reinterpret_cast<const ushort4*>(hp);  // 8B aligned
    float f0 = __uint_as_float((unsigned)raw.x << 16) * w;
    float f1 = __uint_as_float((unsigned)raw.y << 16) * w;
    float f2 = __uint_as_float((unsigned)raw.z << 16) * w;
    float f3 = __uint_as_float((unsigned)raw.w << 16) * w;

    float* d = agg + (size_t)dst * 128 + lane * 4;
    unsafeAtomicAdd(d + 0, f0);   // -> global_atomic_add_f32 (agg fits in L2)
    unsafeAtomicAdd(d + 1, f1);
    unsafeAtomicAdd(d + 2, f2);
    unsafeAtomicAdd(d + 3, f3);
}

// ---------------------------------------------------------------------------
// Epilogues
// ---------------------------------------------------------------------------
__global__ __launch_bounds__(256) void node_relu_kernel(
    const float* __restrict__ agg, const __bf16* __restrict__ Hroot,
    const float* __restrict__ bias, float* __restrict__ x1, int n)
{
    int i = blockIdx.x * 256 + threadIdx.x;
    if (i >= n) return;
    int c = i & 127;
    float v = agg[i] + (float)Hroot[i] + bias[c];
    x1[i] = fmaxf(v, 0.0f);
}

__global__ __launch_bounds__(256) void node_out_kernel(
    const float* __restrict__ agg, const __bf16* __restrict__ Hroot,
    const float* __restrict__ bias, float* __restrict__ out, int n)
{
    int i = blockIdx.x * 256 + threadIdx.x;
    if (i >= n) return;
    int row = i >> 7;
    int c   = i & 127;
    out[(size_t)row * 896 + 768 + c] = agg[i] + (float)Hroot[i] + bias[c];
}

__global__ __launch_bounds__(256) void copy_feat_kernel(
    const float4* __restrict__ f, float* __restrict__ out, int n)  // n = N*192
{
    int i = blockIdx.x * 256 + threadIdx.x;
    if (i >= n) return;
    int row = i / 192;
    int c4  = i % 192;
    reinterpret_cast<float4*>(out + (size_t)row * 896)[c4] = f[i];
}

// ---------------------------------------------------------------------------
static inline size_t align256(size_t x) { return (x + 255) & ~(size_t)255; }

extern "C" void kernel_launch(void* const* d_in, const int* in_sizes, int n_in,
                              void* d_out, int out_size, void* d_ws, size_t ws_size,
                              hipStream_t stream)
{
    const float* feature = (const float*)d_in[0];
    const int*   ei      = (const int*)  d_in[1];
    const int*   et      = (const int*)  d_in[2];
    const float* w1      = (const float*)d_in[3];
    const float* root1   = (const float*)d_in[4];
    const float* b1      = (const float*)d_in[5];
    const float* w2      = (const float*)d_in[6];
    const float* root2   = (const float*)d_in[7];
    const float* b2      = (const float*)d_in[8];
    float* out = (float*)d_out;

    const int nN = in_sizes[0] / 768;   // 50000
    const int nE = in_sizes[2];         // 800000

    // workspace carve (~168 MB)
    char* ws = (char*)d_ws;
    size_t off = 0;
    float*  cnt = (float*)(ws + off);  off += align256((size_t)nN * RREL * 4);
    __bf16* H   = (__bf16*)(ws + off); off += align256((size_t)(RREL + 1) * nN * 128 * 2);
    float*  agg = (float*)(ws + off);  off += align256((size_t)nN * 128 * 4);
    float*  x1  = (float*)(ws + off);

    const int nNC   = nN * 128;
    const int eThr  = nE * 32;
    dim3 gGemm((nN + 63) / 64, RREL + 1);

    // --- degree normalization (shared by both layers) ---
    hipMemsetAsync(cnt, 0, (size_t)nN * RREL * 4, stream);
    edge_count_kernel<<<(nE + 255) / 256, 256, 0, stream>>>(ei, et, (unsigned*)cnt, nE);
    invnorm_kernel<<<(nN * RREL + 255) / 256, 256, 0, stream>>>(cnt, nN * RREL);

    // --- layer 1 ---
    hipMemsetAsync(agg, 0, (size_t)nNC * 4, stream);
    rgemm_bf16_kernel<<<gGemm, 256, 0, stream>>>(feature, w1, root1, H, nN, 768);
    edge_scatter_kernel<<<(eThr + 255) / 256, 256, 0, stream>>>(ei, et, cnt, H, agg, nE, nN);
    node_relu_kernel<<<(nNC + 255) / 256, 256, 0, stream>>>(
        agg, H + (size_t)RREL * nN * 128, b1, x1, nNC);

    // --- layer 2 ---
    hipMemsetAsync(agg, 0, (size_t)nNC * 4, stream);
    rgemm_bf16_kernel<<<gGemm, 256, 0, stream>>>(x1, w2, root2, H, nN, 128);
    edge_scatter_kernel<<<(eThr + 255) / 256, 256, 0, stream>>>(ei, et, cnt, H, agg, nE, nN);
    node_out_kernel<<<(nNC + 255) / 256, 256, 0, stream>>>(
        agg, H + (size_t)RREL * nN * 128, b2, out, nNC);

    // --- concat: out[:, :768] = feature ---
    copy_feat_kernel<<<(nN * 192 + 255) / 256, 256, 0, stream>>>(
        (const float4*)feature, out, nN * 192);
}
